// Conv2d_Parallel_1219770712455
// MI455X (gfx1250) — matrix operand order repaired and checked by reading, not compiled, not run on hardware
//
#include <hip/hip_runtime.h>
#include <hip/hip_bf16.h>

typedef __attribute__((ext_vector_type(16))) __bf16 v16bf;
typedef __attribute__((ext_vector_type(8)))  float  v8f;

#define NPATCH 40000
#define G 4
#define K 64
#define D 400
#define DPAD 416          // D padded to multiple of 32 for WMMA K-loop
#define LROW 424          // LDS row stride (halves): 212 dwords, 212%64=20 -> conflict-free
#define EPOCHS 10
#define NTILES_B 20       // 40000 / 2000

// pack two f32 -> two bf16 (round-half-up) in one v_perm_b32
static __device__ __forceinline__ unsigned pk2bf(float a, float b) {
    union { float f; unsigned u; } x, y; x.f = a; y.f = b;
    const unsigned ua = x.u + 0x8000u;
    const unsigned ub = y.u + 0x8000u;
    // dst = { ub.b3, ub.b2, ua.b3, ua.b2 }  (selector bytes 7,6,3,2; S0=ub, S1=ua)
    return __builtin_amdgcn_perm(ub, ua, 0x07060302u);
}

static __device__ __forceinline__ unsigned short f2bf(float f) {   // RNE, cold paths
    union { float f; unsigned u; } x; x.f = f;
    unsigned r = x.u + 0x7FFFu + ((x.u >> 16) & 1u);
    return (unsigned short)(r >> 16);
}

// ---------------------------------------------------------------------------
// Kernel 1: score GEMM (WMMA bf16) + argmax -> labels
// grid: (313, 4)  block: 256 (8 waves, 16 patches/wave -> 128 patches/block)
// ---------------------------------------------------------------------------
__global__ __launch_bounds__(256)
void kmeans_assign(const float* __restrict__ patches,
                   const unsigned short* __restrict__ centB,   // [G][K][DPAD] bf16
                   const float* __restrict__ c2,               // [G][K]
                   unsigned* __restrict__ labels)              // [G][N]
{
    __shared__ __align__(32) unsigned short lcent[K * LROW];   // 54,272 B
    const int g   = blockIdx.y;
    const int tid = threadIdx.x;

    // stage centroids (bf16) into LDS, padded row stride
    {
        const unsigned short* src = centB + (size_t)g * K * DPAD;
        for (int i = tid; i < K * DPAD; i += 256) {
            int row = i / DPAD, pos = i - row * DPAD;
            lcent[row * LROW + pos] = src[i];
        }
    }
    __syncthreads();

    const int wave = tid >> 5;
    const int lane = tid & 31;
    const int half = lane >> 4;
    const int l16  = lane & 15;
    const int nbase = blockIdx.x * 128 + wave * 16;

    int nrow = nbase + l16;                 // patch row this lane supplies for A
    if (nrow > NPATCH - 1) nrow = NPATCH - 1;
    const float* prow = patches + ((size_t)nrow * G + g) * D;

    float c2v[4];
#pragma unroll
    for (int c = 0; c < 4; ++c) c2v[c] = c2[g * K + c * 16 + l16];

    v8f acc[4] = {};

    // ---- main K-loop: 12 branch-free steps cover d = 0..383 ----
    for (int kb = 0; kb < 384; kb += 32) {
        const int d0 = kb + 16 * half;         // elems 0..15 : K = 16*half + j
        const float4 x0 = *(const float4*)(prow + d0);
        const float4 x1 = *(const float4*)(prow + d0 + 4);
        const float4 x2 = *(const float4*)(prow + d0 + 8);
        const float4 x3 = *(const float4*)(prow + d0 + 12);
        if (d0 + 96 < D) __builtin_prefetch(prow + d0 + 96, 0, 3);

        union { v16bf v; unsigned w[8]; } au;
        au.w[0] = pk2bf(x0.x, x0.y); au.w[1] = pk2bf(x0.z, x0.w);
        au.w[2] = pk2bf(x1.x, x1.y); au.w[3] = pk2bf(x1.z, x1.w);
        au.w[4] = pk2bf(x2.x, x2.y); au.w[5] = pk2bf(x2.z, x2.w);
        au.w[6] = pk2bf(x3.x, x3.y); au.w[7] = pk2bf(x3.z, x3.w);

        const int boff = kb + 16 * half;       // contiguous 16 halves (32B)
#pragma unroll
        for (int c = 0; c < 4; ++c) {
            const v16bf bv = *(const v16bf*)(lcent + (c * 16 + l16) * LROW + boff);
            acc[c] = __builtin_amdgcn_wmma_f32_16x16x32_bf16(
                false, au.v, false, bv, (short)0, acc[c], false, false);
        }
    }

    // ---- peeled tail: kb = 384, d = 384..399 valid, 400..415 are zeros ----
    {
        const int d0 = 384;                    // half 0: d 384..399; half 1: zeros (400..415)
        const float4 x0 = *(const float4*)(prow + d0);
        const float4 x1 = *(const float4*)(prow + d0 + 4);
        const float4 x2 = *(const float4*)(prow + d0 + 8);
        const float4 x3 = *(const float4*)(prow + d0 + 12);
        union { v16bf v; unsigned w[8]; } au;
        au.w[0] = half ? 0u : pk2bf(x0.x, x0.y); au.w[1] = half ? 0u : pk2bf(x0.z, x0.w);
        au.w[2] = half ? 0u : pk2bf(x1.x, x1.y); au.w[3] = half ? 0u : pk2bf(x1.z, x1.w);
        au.w[4] = half ? 0u : pk2bf(x2.x, x2.y); au.w[5] = half ? 0u : pk2bf(x2.z, x2.w);
        au.w[6] = half ? 0u : pk2bf(x3.x, x3.y); au.w[7] = half ? 0u : pk2bf(x3.z, x3.w);

        const int boff = 384 + 16 * half;      // LDS pad region holds zeros
#pragma unroll
        for (int c = 0; c < 4; ++c) {
            const v16bf bv = *(const v16bf*)(lcent + (c * 16 + l16) * LROW + boff);
            acc[c] = __builtin_amdgcn_wmma_f32_16x16x32_bf16(
                false, au.v, false, bv, (short)0, acc[c], false, false);
        }
    }

    // ---- argmax over 64 centroids per patch ----
    // C layout: reg r, lane -> M = r + 8*half, N(col) = l16; k = c*16 + l16
#pragma unroll
    for (int r = 0; r < 8; ++r) {
        float bv = acc[0][r] - c2v[0];
        int   bk = 0 * 16 + l16;
#pragma unroll
        for (int c = 1; c < 4; ++c) {
            float v = acc[c][r] - c2v[c];
            int   kk = c * 16 + l16;
            if (v > bv || (v == bv && kk < bk)) { bv = v; bk = kk; }
        }
        // reduce across the 16 lanes of this half (xor 1,2,4,8 stays in-half)
#pragma unroll
        for (int s = 8; s >= 1; s >>= 1) {
            float ov = __shfl_xor(bv, s, 32);
            int   ok = __shfl_xor(bk, s, 32);
            if (ov > bv || (ov == bv && ok < bk)) { bv = ov; bk = ok; }
        }
        const int m  = r + 8 * half;
        const int nn = nbase + m;
        if (l16 == 0 && nn < NPATCH)
            labels[g * NPATCH + nn] = (unsigned)bk;
    }
}

// ---------------------------------------------------------------------------
// Kernel 2: deterministic partial accumulation (no atomics).
// grid: (20, 2, 4) = (tile, d-half, group); block: 256. Thread t owns d=dbase+t.
// ---------------------------------------------------------------------------
__global__ __launch_bounds__(256)
void kmeans_accum(const float* __restrict__ patches,
                  const unsigned* __restrict__ labels,
                  float* __restrict__ partial,        // [G][20][K][D]
                  unsigned* __restrict__ counts_p)    // [G][20][K]
{
    __shared__ float    lacc[K * 200];                // 51,200 B
    __shared__ unsigned lcnt[K];
    const int tile = blockIdx.x;
    const int dh   = blockIdx.y;
    const int g    = blockIdx.z;
    const int t    = threadIdx.x;

    for (int i = t; i < K * 200; i += 256) lacc[i] = 0.f;
    if (t < K) lcnt[t] = 0u;
    __syncthreads();

    const int dbase = dh * 200;
    const int n0 = tile * 2000;
    if (t < 200) {
        for (int n = n0; n < n0 + 2000; ++n) {
            const unsigned lab = labels[g * NPATCH + n];
            const float v = patches[((size_t)n * G + g) * D + dbase + t];
            if (n + 4 < n0 + 2000)
                __builtin_prefetch(&patches[((size_t)(n + 4) * G + g) * D + dbase + t], 0, 3);
            lacc[lab * 200 + t] += v;            // column t exclusively owned
            if (t == 0 && dh == 0) lcnt[lab] += 1u;
        }
    }
    __syncthreads();

    float* pout = partial + (((size_t)g * NTILES_B + tile) * K) * D;
    for (int i = t; i < K * 200; i += 256) {
        const int k = i / 200, d = i - k * 200;
        pout[(size_t)k * D + dbase + d] = lacc[i];
    }
    if (dh == 0 && t < K)
        counts_p[(g * NTILES_B + tile) * K + t] = lcnt[t];
}

// ---------------------------------------------------------------------------
// Kernel 3: reduce counts (fixed order), empty-cluster mask, per-row scale.
// grid: 1, block: 256 (thread = g*64+k)
// ---------------------------------------------------------------------------
__global__ __launch_bounds__(256)
void kmeans_scale(const unsigned* __restrict__ counts_p,
                  float* __restrict__ scale)          // [G*K]
{
    __shared__ unsigned scnt[G * K];
    const int t = threadIdx.x;
    const int g = t >> 6, k = t & 63;
    unsigned c = 0;
    for (int tile = 0; tile < NTILES_B; ++tile)
        c += counts_p[(g * NTILES_B + tile) * K + k];
    scnt[t] = c;
    __syncthreads();
    const bool bad = (scnt[k] == 0u) | (scnt[K + k] == 0u) |
                     (scnt[2 * K + k] == 0u) | (scnt[3 * K + k] == 0u);
    scale[t] = bad ? 0.f : 1.f / (float)(c == 0u ? 1u : c);
}

// ---------------------------------------------------------------------------
// Kernel 4: fixed-order tile reduce, update centroids (f32 out + bf16 + c2).
// grid: 256 (one block per (g,k) row), block: 256
// ---------------------------------------------------------------------------
__global__ __launch_bounds__(256)
void kmeans_update(const float* __restrict__ partial,
                   const float* __restrict__ scale,
                   unsigned short* __restrict__ centB,  // [G][K][DPAD]
                   float* __restrict__ c2,
                   float* __restrict__ out)             // [G*K][D]
{
    __shared__ float red[256];
    const int row = blockIdx.x;          // g*64+k
    const int g = row >> 6, k = row & 63;
    const int t = threadIdx.x;
    const float s = scale[row];
    float sq = 0.f;
    for (int d = t; d < D; d += 256) {
        float acc = 0.f;
        for (int tile = 0; tile < NTILES_B; ++tile)
            acc += partial[(((size_t)g * NTILES_B + tile) * K + k) * D + d];
        const float val = acc * s;
        sq += val * val;
        out[(size_t)row * D + d] = val;
        centB[(size_t)row * DPAD + d] = f2bf(val);
    }
    red[t] = sq; __syncthreads();
    for (int off = 128; off >= 1; off >>= 1) {
        if (t < off) red[t] += red[t + off];
        __syncthreads();
    }
    if (t == 0) c2[row] = 0.5f * red[0];
}

// ---------------------------------------------------------------------------
// Prep: init centroids -> bf16 (zero-padded) + c2.  grid: 256, block: 256
// ---------------------------------------------------------------------------
__global__ __launch_bounds__(256)
void kmeans_prep(const float* __restrict__ cinit,
                 unsigned short* __restrict__ centB,
                 float* __restrict__ c2)
{
    __shared__ float red[256];
    const int row = blockIdx.x;
    const int t = threadIdx.x;
    float sq = 0.f;
    for (int d = t; d < DPAD; d += 256) {
        const float v = (d < D) ? cinit[(size_t)row * D + d] : 0.f;
        centB[(size_t)row * DPAD + d] = f2bf(v);
        if (d < D) sq += v * v;
    }
    red[t] = sq; __syncthreads();
    for (int off = 128; off >= 1; off >>= 1) {
        if (t < off) red[t] += red[t + off];
        __syncthreads();
    }
    if (t == 0) c2[row] = 0.5f * red[0];
}

extern "C" void kernel_launch(void* const* d_in, const int* in_sizes, int n_in,
                              void* d_out, int out_size, void* d_ws, size_t ws_size,
                              hipStream_t stream) {
    const float* patches = (const float*)d_in[0];   // [40000][4][400]
    const float* cinit   = (const float*)d_in[1];   // [4][64][400]
    float* out = (float*)d_out;                     // [256][400]

    char* ws = (char*)d_ws;
    // workspace carve (bytes):
    unsigned short* centB    = (unsigned short*)(ws);              // 256*416*2 = 212,992
    float*          c2       = (float*)(ws + 212992);              // 1,024
    float*          scale    = (float*)(ws + 214016);              // 1,024
    unsigned*       labels   = (unsigned*)(ws + 215040);           // 4*40000*4 = 640,000
    unsigned*       counts_p = (unsigned*)(ws + 855040);           // 4*20*64*4 = 20,480
    float*          partial  = (float*)(ws + 875520);              // 4*20*64*400*4 = 8,192,000
    (void)in_sizes; (void)n_in; (void)out_size; (void)ws_size;

    kmeans_prep<<<256, 256, 0, stream>>>(cinit, centB, c2);
    for (int e = 0; e < EPOCHS; ++e) {
        kmeans_assign<<<dim3(313, G), 256, 0, stream>>>(patches, centB, c2, labels);
        kmeans_accum<<<dim3(NTILES_B, 2, G), 256, 0, stream>>>(patches, labels, partial, counts_p);
        kmeans_scale<<<1, 256, 0, stream>>>(counts_p, scale);
        kmeans_update<<<256, 256, 0, stream>>>(partial, scale, centB, c2, out);
    }
}
